// QCConv_49761491092014
// MI455X (gfx1250) — matrix operand order repaired and checked by reading, not compiled, not run on hardware
//
#include <hip/hip_runtime.h>
#include <math.h>

#define cN 100000
#define cE 500000
#define cD 128
#define cD2 256
#define cH 4
#define EPSF 1e-5f
#define SCALEF 0.0625f   // 1/sqrt(2*D) = 1/16
#define NSLOT 64

typedef __attribute__((ext_vector_type(16))) __bf16 v16bf;
typedef __attribute__((ext_vector_type(8)))  float  v8f;

// ---------------- scalar helpers ----------------
__device__ static inline unsigned short f2bf(float f) {
    unsigned int u = __float_as_uint(f);
    u += 0x7FFFu + ((u >> 16) & 1u);          // round-to-nearest-even
    return (unsigned short)(u >> 16);
}
__device__ static inline float bf2f(unsigned short h) {
    return __uint_as_float(((unsigned int)h) << 16);
}
__device__ static inline float siluf(float x) { return x / (1.f + __expf(-x)); }
__device__ static inline float sigmf(float x) { return 1.f / (1.f + __expf(-x)); }

__device__ static inline v8f zero8() {
    v8f z;
#pragma unroll
    for (int i = 0; i < 8; ++i) z[i] = 0.f;
    return z;
}

__device__ static inline v8f wmma_bf(v16bf a, v16bf b, v8f c) {
    return __builtin_amdgcn_wmma_f32_16x16x32_bf16(false, a, false, b,
                                                   (short)0, c, false, false);
}

// ---------------- async memory->LDS (per-lane 16B DMA, ASYNCcnt) ----------------
// LDS address comes from the VDST VGPR; generic pointer low 32 bits == LDS byte offset.
__device__ static inline void async_ld16(void* lds_ptr, const void* gptr) {
    unsigned lds_off = (unsigned)(unsigned long long)lds_ptr;
    asm volatile("global_load_async_to_lds_b128 %0, %1, off"
                 :: "v"(lds_off), "v"(gptr)
                 : "memory");
}
__device__ static inline void async_wait0() {
    asm volatile("s_wait_asynccnt 0x0" ::: "memory");
}

// A fragment (16x32 bf16, M x K) from LDS row-major [rows][stride] ushort.
// ISA layout: lane L holds row M = L%16; element j holds
//   K = kbase + (j%8) + 16*(j/8) + 8*(L>=16)
__device__ static inline v16bf lds_a(const unsigned short* A, int stride,
                                     int mbase, int kbase, int lane) {
    const unsigned short* p = A + (mbase + (lane & 15)) * stride + kbase + ((lane >> 4) << 3);
    union { unsigned int u[8]; v16bf v; } r;
#pragma unroll
    for (int g = 0; g < 2; ++g)
#pragma unroll
        for (int q = 0; q < 4; ++q)
            r.u[g * 4 + q] = *(const unsigned int*)(p + g * 16 + q * 2);
    return r.v;
}

// B fragment (32x16 bf16, K x N) from pre-packed global weights:
// packed[((n_tile*nK + ks)*32 + lane)*16 + j]; one 32B contiguous load/lane.
__device__ static inline v16bf gl_b(const unsigned short* pk, int n_tile,
                                    int ks, int nK, int lane) {
    const uint4* p = (const uint4*)(pk + ((((size_t)n_tile * nK + ks) * 32 + lane) << 4));
    union { uint4 q[2]; v16bf v; } r;
    r.q[0] = p[0]; r.q[1] = p[1];
    return r.v;
}

// ---------------- weight packing ----------------
// src: f32 row-major [h][K][Nc]; dst: bf16 fragment-major, same elem count.
// B layout: element j of lane L: K = ks*32 + (j%8) + 16*(j/8) + 8*(L>=16), N = n_tile*16 + L%16
__global__ __launch_bounds__(256) void pack_b_kernel(const float* src,
                                                     unsigned short* dst,
                                                     int K, int Nc) {
    int h = blockIdx.y;
    const float* S = src + (size_t)h * K * Nc;
    unsigned short* Dp = dst + (size_t)h * K * Nc;
    int total = K * Nc;
    int nK = K >> 5;
    for (int t = blockIdx.x * blockDim.x + threadIdx.x; t < total;
         t += gridDim.x * blockDim.x) {
        int j = t & 15;
        int lane = (t >> 4) & 31;
        int rest = t >> 9;
        int ks = rest % nK;
        int n_tile = rest / nK;
        int Kof = ks * 32 + (j & 7) + ((j >> 3) << 4) + ((lane >> 4) << 3);
        int n = n_tile * 16 + (lane & 15);
        Dp[t] = f2bf(S[(size_t)Kof * Nc + n]);
    }
}

__global__ __launch_bounds__(256) void cvt_bf_kernel(const float* s,
                                                     unsigned short* d,
                                                     long long n) {
    for (long long i = (long long)blockIdx.x * blockDim.x + threadIdx.x; i < n;
         i += (long long)gridDim.x * blockDim.x)
        d[i] = f2bf(s[i]);
}

// ---------------- node projection GEMM: [N,128] @ [128,128] -> bf16 ----------------
__global__ __launch_bounds__(256) void gemm_node_kernel(const unsigned short* xbf,
                                                        const unsigned short* pkK,
                                                        const unsigned short* pkV,
                                                        unsigned short* outK,
                                                        unsigned short* outV) {
    __shared__ __align__(16) unsigned short As[32 * 128];
    int h = blockIdx.y & 3;
    int isV = blockIdx.y >> 2;
    const unsigned short* pk = (isV ? pkV : pkK) + (size_t)h * cD * cD;
    unsigned short* out = (isV ? outV : outK) + (size_t)h * cN * cD;
    int r0 = blockIdx.x * 32;
    int tid = threadIdx.x, lane = tid & 31, wave = tid >> 5;
    // stage 32 rows of x (already bf16) via async DMA: 512 x 16B chunks
    const char* src = (const char*)(xbf + (size_t)r0 * cD);
#pragma unroll
    for (int t = 0; t < 2; ++t) {
        int i = tid + t * 256;
        async_ld16((char*)As + i * 16, src + i * 16);
    }
    async_wait0();
    __syncthreads();
    int n_tile = wave;                   // 8 waves -> 8 n-tiles of 16
    v8f a0 = zero8(), a1 = zero8();
#pragma unroll
    for (int ks = 0; ks < 4; ++ks) {
        v16bf b = gl_b(pk, n_tile, ks, 4, lane);
        a0 = wmma_bf(lds_a(As, 128, 0, ks * 32, lane), b, a0);
        a1 = wmma_bf(lds_a(As, 128, 16, ks * 32, lane), b, a1);
    }
    int col = n_tile * 16 + (lane & 15);
    int hi8 = (lane >> 4) << 3;
#pragma unroll
    for (int r = 0; r < 8; ++r) {
        out[(size_t)(r0 + r + hi8) * cD + col] = f2bf(a0[r]);
        out[(size_t)(r0 + 16 + r + hi8) * cD + col] = f2bf(a1[r]);
    }
}

// ---------------- shared device pieces for edge kernels ----------------
// 128-wide projection of EF tile into Out[:,128:256] (stride 256)
__device__ static inline void proj128(const unsigned short* EF,
                                      const unsigned short* pk,
                                      unsigned short* Out, int wave, int lane) {
    int n_tile = wave;
    v8f a0 = zero8(), a1 = zero8();
#pragma unroll
    for (int ks = 0; ks < 4; ++ks) {
        v16bf b = gl_b(pk, n_tile, ks, 4, lane);
        a0 = wmma_bf(lds_a(EF, 128, 0, ks * 32, lane), b, a0);
        a1 = wmma_bf(lds_a(EF, 128, 16, ks * 32, lane), b, a1);
    }
    int col = 128 + n_tile * 16 + (lane & 15);
    int hi8 = (lane >> 4) << 3;
#pragma unroll
    for (int r = 0; r < 8; ++r) {
        Out[(r + hi8) * 256 + col] = f2bf(a0[r]);
        Out[(16 + r + hi8) * 256 + col] = f2bf(a1[r]);
    }
}

// 256x256 layer: Out = act(Ain @ pk + bias), bf16 out, stride 256, 32 rows
__device__ static inline void layer256(const unsigned short* Ain,
                                       const unsigned short* pk,
                                       const float* bias, unsigned short* Out,
                                       int do_silu, int wave, int lane) {
    int hi8 = (lane >> 4) << 3;
#pragma unroll
    for (int t = 0; t < 2; ++t) {
        int n_tile = wave + t * 8;
        v8f a0 = zero8(), a1 = zero8();
#pragma unroll
        for (int ks = 0; ks < 8; ++ks) {
            v16bf b = gl_b(pk, n_tile, ks, 8, lane);
            a0 = wmma_bf(lds_a(Ain, 256, 0, ks * 32, lane), b, a0);
            a1 = wmma_bf(lds_a(Ain, 256, 16, ks * 32, lane), b, a1);
        }
        int col = n_tile * 16 + (lane & 15);
        float bc = bias[col];
#pragma unroll
        for (int r = 0; r < 8; ++r) {
            float v0 = a0[r] + bc, v1 = a1[r] + bc;
            if (do_silu) { v0 = siluf(v0); v1 = siluf(v1); }
            Out[(r + hi8) * 256 + col] = f2bf(v0);
            Out[(16 + r + hi8) * 256 + col] = f2bf(v1);
        }
    }
}

// Gather 32 rows (256B each) from rows[] of table into LDS (stride 256 ushorts,
// columns 0..127) via async DMA. 512 chunks of 16B over 256 threads.
__device__ static inline void async_gather_rows(unsigned short* dstLDS,
                                                const unsigned short* table,
                                                const int* rows, int tid) {
#pragma unroll
    for (int t = 0; t < 2; ++t) {
        int i = tid + t * 256;
        int r = i >> 4;            // row 0..31
        int c = i & 15;            // 16B chunk 0..15 (=> 8 ushorts)
        const char* g = (const char*)(table + (size_t)rows[r] * cD + c * 8);
        async_ld16((char*)(dstLDS + r * 256 + c * 8), g);
    }
}

// ---------------- edge pass 1: alpha statistics ----------------
__global__ __launch_bounds__(256) void edge_pass1_kernel(
    const float* ef, const int* eidx, const unsigned short* Kv_bf,
    const unsigned short* pkKe, const unsigned short* pkKu1,
    const unsigned short* pkKu2, const float* ku_b1, const float* ku_b2,
    float* bn_part) {
    __shared__ __align__(16) unsigned short EF[32 * 128];
    __shared__ __align__(16) unsigned short A[32 * 256];
    __shared__ __align__(16) unsigned short Hd[32 * 256];
    __shared__ float stats[512];
    __shared__ int sE[32], dE[32];
    int h = blockIdx.y;
    int e0 = blockIdx.x * 32;
    int tid = threadIdx.x, lane = tid & 31, wave = tid >> 5;
    int hi8 = (lane >> 4) << 3;
    if (tid < 32) { sE[tid] = eidx[e0 + tid]; dE[tid] = eidx[cE + e0 + tid]; }
    for (int i = tid; i < 512; i += 256) stats[i] = 0.f;
    __syncthreads();
    // async gather Kv[src] rows into A left half; overlap with EF f32->bf16
    async_gather_rows(A, Kv_bf + (size_t)h * cN * cD, sE, tid);
    for (int i = tid; i < 32 * 128; i += 256) {
        int r = i >> 7, c = i & 127;
        EF[i] = f2bf(ef[(size_t)(e0 + r) * cD + c]);
    }
    async_wait0();
    __syncthreads();
    proj128(EF, pkKe + (size_t)h * cD * cD, A, wave, lane);   // KE into A[:,128:]
    __syncthreads();
    layer256(A, pkKu1 + (size_t)h * cD2 * cD2, ku_b1 + h * cD2, Hd, 1, wave, lane);
    __syncthreads();
    {   // layer 2 + alpha stats
        const unsigned short* pk = pkKu2 + (size_t)h * cD2 * cD2;
        const float* bias = ku_b2 + h * cD2;
#pragma unroll
        for (int t = 0; t < 2; ++t) {
            int n_tile = wave + t * 8;
            v8f a0 = zero8(), a1 = zero8();
#pragma unroll
            for (int ks = 0; ks < 8; ++ks) {
                v16bf b = gl_b(pk, n_tile, ks, 8, lane);
                a0 = wmma_bf(lds_a(Hd, 256, 0, ks * 32, lane), b, a0);
                a1 = wmma_bf(lds_a(Hd, 256, 16, ks * 32, lane), b, a1);
            }
            int col = n_tile * 16 + (lane & 15);
            float bc = bias[col];
            float s = 0.f, sq = 0.f;
#pragma unroll
            for (int r = 0; r < 8; ++r) {
                int row0 = r + hi8, row1 = 16 + r + hi8;
                float k0 = a0[r] + bc, k1 = a1[r] + bc;
                float q0 = bf2f(Kv_bf[((size_t)h * cN + dE[row0]) * cD + (col & 127)]);
                float q1 = bf2f(Kv_bf[((size_t)h * cN + dE[row1]) * cD + (col & 127)]);
                float al0 = q0 * k0 * SCALEF, al1 = q1 * k1 * SCALEF;
                s += al0 + al1;
                sq += al0 * al0 + al1 * al1;
            }
            atomicAdd(&stats[col], s);
            atomicAdd(&stats[256 + col], sq);
        }
    }
    __syncthreads();
    int slot = blockIdx.x & (NSLOT - 1);
    {
        int i = tid;   // 256 threads, 256 features
        atomicAdd(&bn_part[(((size_t)h * NSLOT + slot) * 2 + 0) * cD2 + i], stats[i]);
        atomicAdd(&bn_part[(((size_t)h * NSLOT + slot) * 2 + 1) * cD2 + i], stats[256 + i]);
    }
}

__global__ void reduce_bn_kernel(const float* part, const float* g,
                                 const float* b, float* stat) {
    int h = blockIdx.x, t = threadIdx.x;   // 256 threads
    float s = 0.f, q = 0.f;
    for (int k = 0; k < NSLOT; ++k) {
        s += part[(((size_t)h * NSLOT + k) * 2 + 0) * cD2 + t];
        q += part[(((size_t)h * NSLOT + k) * 2 + 1) * cD2 + t];
    }
    float mean = s / (float)cE;
    float var = q / (float)cE - mean * mean;
    float istd = rsqrtf(var + EPSF);
    float gg = g[h * cD2 + t], bb = b[h * cD2 + t];
    stat[(h * 2 + 0) * cD2 + t] = istd * gg;            // fused scale
    stat[(h * 2 + 1) * cD2 + t] = bb - mean * istd * gg; // fused bias
}

// ---------------- edge pass 2: full chain + scatter ----------------
__global__ __launch_bounds__(256) void edge_pass2_kernel(
    const float* ef, const int* eidx, const unsigned short* Kv_bf,
    const unsigned short* Vv_bf, const unsigned short* pkKe,
    const unsigned short* pkVe, const unsigned short* pkKu1,
    const unsigned short* pkKu2, const unsigned short* pkLu1,
    const unsigned short* pkLu2, const unsigned short* pkMsg,
    const float* ku_b1, const float* ku_b2, const float* lu_b1,
    const float* lu_b2, const float* msg_b, const float* ln_g,
    const float* ln_b, const float* bn_stat, float* hv) {
    __shared__ __align__(16) unsigned short EF[32 * 128];
    __shared__ __align__(16) unsigned short A[32 * 256];   // k_in -> gate
    __shared__ __align__(16) unsigned short Hd[32 * 256];  // hidden1 -> v_in -> m
    __shared__ __align__(16) float MSGf[32 * 128];         // hidden2(bf16 view) -> msg f32
    __shared__ int sE[32], dE[32];
    unsigned short* H2 = (unsigned short*)MSGf;
    int h = blockIdx.y;
    int e0 = blockIdx.x * 32;
    int tid = threadIdx.x, lane = tid & 31, wave = tid >> 5;
    int hi8 = (lane >> 4) << 3;
    if (tid < 32) { sE[tid] = eidx[e0 + tid]; dE[tid] = eidx[cE + e0 + tid]; }
    __syncthreads();
    // async gather Kv[src] into A left half; overlap with EF conversion
    async_gather_rows(A, Kv_bf + (size_t)h * cN * cD, sE, tid);
    for (int i = tid; i < 32 * 128; i += 256) {
        int r = i >> 7, c = i & 127;
        EF[i] = f2bf(ef[(size_t)(e0 + r) * cD + c]);
    }
    async_wait0();
    __syncthreads();
    proj128(EF, pkKe + (size_t)h * cD * cD, A, wave, lane);      // KE
    __syncthreads();
    layer256(A, pkKu1 + (size_t)h * cD2 * cD2, ku_b1 + h * cD2, Hd, 1, wave, lane);
    __syncthreads();
    {   // ku layer 2 -> gate stored into A
        const unsigned short* pk = pkKu2 + (size_t)h * cD2 * cD2;
        const float* bias = ku_b2 + h * cD2;
        const float* gs = bn_stat + (size_t)h * 2 * cD2;
        const float* gb = gs + cD2;
#pragma unroll
        for (int t = 0; t < 2; ++t) {
            int n_tile = wave + t * 8;
            v8f a0 = zero8(), a1 = zero8();
#pragma unroll
            for (int ks = 0; ks < 8; ++ks) {
                v16bf b = gl_b(pk, n_tile, ks, 8, lane);
                a0 = wmma_bf(lds_a(Hd, 256, 0, ks * 32, lane), b, a0);
                a1 = wmma_bf(lds_a(Hd, 256, 16, ks * 32, lane), b, a1);
            }
            int col = n_tile * 16 + (lane & 15);
            float bc = bias[col], sc = gs[col], sb = gb[col];
#pragma unroll
            for (int r = 0; r < 8; ++r) {
                int row0 = r + hi8, row1 = 16 + r + hi8;
                float k0 = a0[r] + bc, k1 = a1[r] + bc;
                float q0 = bf2f(Kv_bf[((size_t)h * cN + dE[row0]) * cD + (col & 127)]);
                float q1 = bf2f(Kv_bf[((size_t)h * cN + dE[row1]) * cD + (col & 127)]);
                float g0 = sigmf(q0 * k0 * SCALEF * sc + sb);
                float g1 = sigmf(q1 * k1 * SCALEF * sc + sb);
                A[row0 * 256 + col] = f2bf(g0);
                A[row1 * 256 + col] = f2bf(g1);
            }
        }
    }
    __syncthreads();
    // v_in into Hd: left = Vv[src] (async DMA), right = VE = EF @ V_e2v (overlap)
    async_gather_rows(Hd, Vv_bf + (size_t)h * cN * cD, sE, tid);
    proj128(EF, pkVe + (size_t)h * cD * cD, Hd, wave, lane);
    async_wait0();
    __syncthreads();
    layer256(Hd, pkLu1 + (size_t)h * cD2 * cD2, lu_b1 + h * cD2, H2, 1, wave, lane);
    __syncthreads();
    {   // lu layer 2 -> m = v * gate, stored into Hd
        const unsigned short* pk = pkLu2 + (size_t)h * cD2 * cD2;
        const float* bias = lu_b2 + h * cD2;
#pragma unroll
        for (int t = 0; t < 2; ++t) {
            int n_tile = wave + t * 8;
            v8f a0 = zero8(), a1 = zero8();
#pragma unroll
            for (int ks = 0; ks < 8; ++ks) {
                v16bf b = gl_b(pk, n_tile, ks, 8, lane);
                a0 = wmma_bf(lds_a(H2, 256, 0, ks * 32, lane), b, a0);
                a1 = wmma_bf(lds_a(H2, 256, 16, ks * 32, lane), b, a1);
            }
            int col = n_tile * 16 + (lane & 15);
            float bc = bias[col];
#pragma unroll
            for (int r = 0; r < 8; ++r) {
                int row0 = r + hi8, row1 = 16 + r + hi8;
                float v0 = (a0[r] + bc) * bf2f(A[row0 * 256 + col]);
                float v1 = (a1[r] + bc) * bf2f(A[row1 * 256 + col]);
                Hd[row0 * 256 + col] = f2bf(v0);
                Hd[row1 * 256 + col] = f2bf(v1);
            }
        }
    }
    __syncthreads();
    {   // msg layer: [32,256] @ [256,128] -> MSGf (f32)
        const unsigned short* pk = pkMsg + (size_t)h * cD2 * cD;
        const float* bias = msg_b + h * cD;
        int n_tile = wave;
        v8f a0 = zero8(), a1 = zero8();
#pragma unroll
        for (int ks = 0; ks < 8; ++ks) {
            v16bf b = gl_b(pk, n_tile, ks, 8, lane);
            a0 = wmma_bf(lds_a(Hd, 256, 0, ks * 32, lane), b, a0);
            a1 = wmma_bf(lds_a(Hd, 256, 16, ks * 32, lane), b, a1);
        }
        int col = n_tile * 16 + (lane & 15);
        float bc = bias[col];
#pragma unroll
        for (int r = 0; r < 8; ++r) {
            MSGf[(r + hi8) * 128 + col] = a0[r] + bc;
            MSGf[(16 + r + hi8) * 128 + col] = a1[r] + bc;
        }
    }
    __syncthreads();
    if (tid < 32) {   // LayerNorm + SiLU + scatter-add
        int row = tid;
        float mu = 0.f;
        for (int f = 0; f < cD; ++f) mu += MSGf[row * 128 + f];
        mu *= (1.f / cD);
        float var = 0.f;
        for (int f = 0; f < cD; ++f) {
            float d = MSGf[row * 128 + f] - mu;
            var += d * d;
        }
        var *= (1.f / cD);
        float istd = rsqrtf(var + EPSF);
        const float* lg = ln_g + h * cD;
        const float* lb = ln_b + h * cD;
        size_t base = (size_t)dE[row] * (cH * cD) + (size_t)h * cD;
        for (int f = 0; f < cD; ++f) {
            float v = (MSGf[row * 128 + f] - mu) * istd * lg[f] + lb[f];
            atomicAdd(&hv[base + f], siluf(v));
        }
    }
}

// ---------------- concat GEMM: [N,512] @ [512,128] + stats ----------------
__global__ __launch_bounds__(256) void cat_gemm_kernel(const float* hv,
                                                       const unsigned short* pkCat,
                                                       const float* cat_b, float* y,
                                                       float* bnv_part) {
    __shared__ __align__(16) unsigned short As[32 * 512];
    __shared__ float stats[256];
    int r0 = blockIdx.x * 32;
    int tid = threadIdx.x, lane = tid & 31, wave = tid >> 5;
    int hi8 = (lane >> 4) << 3;
    if (tid < 256) stats[tid] = 0.f;
    for (int i = tid; i < 32 * 512; i += 256)
        As[i] = f2bf(hv[(size_t)r0 * 512 + i]);
    __syncthreads();
    int n_tile = wave;
    v8f a0 = zero8(), a1 = zero8();
#pragma unroll
    for (int ks = 0; ks < 16; ++ks) {
        v16bf b = gl_b(pkCat, n_tile, ks, 16, lane);
        a0 = wmma_bf(lds_a(As, 512, 0, ks * 32, lane), b, a0);
        a1 = wmma_bf(lds_a(As, 512, 16, ks * 32, lane), b, a1);
    }
    int col = n_tile * 16 + (lane & 15);
    float bc = cat_b[col];
    float s = 0.f, sq = 0.f;
#pragma unroll
    for (int r = 0; r < 8; ++r) {
        float v0 = a0[r] + bc, v1 = a1[r] + bc;
        y[(size_t)(r0 + r + hi8) * cD + col] = v0;
        y[(size_t)(r0 + 16 + r + hi8) * cD + col] = v1;
        s += v0 + v1;
        sq += v0 * v0 + v1 * v1;
    }
    atomicAdd(&stats[col], s);
    atomicAdd(&stats[128 + col], sq);
    __syncthreads();
    int slot = blockIdx.x & (NSLOT - 1);
    if (tid < 128) {
        atomicAdd(&bnv_part[((size_t)slot * 2 + 0) * cD + tid], stats[tid]);
        atomicAdd(&bnv_part[((size_t)slot * 2 + 1) * cD + tid], stats[128 + tid]);
    }
}

__global__ void reduce_bnv_kernel(const float* part, const float* g,
                                  const float* b, float* stat) {
    int t = threadIdx.x;   // 128 threads
    float s = 0.f, q = 0.f;
    for (int k = 0; k < NSLOT; ++k) {
        s += part[((size_t)k * 2 + 0) * cD + t];
        q += part[((size_t)k * 2 + 1) * cD + t];
    }
    float mean = s / (float)cN;
    float var = q / (float)cN - mean * mean;
    float istd = rsqrtf(var + EPSF);
    float gg = g[t], bb = b[t];
    stat[t] = istd * gg;
    stat[cD + t] = bb - mean * istd * gg;
}

__global__ __launch_bounds__(256) void final_kernel(const float* x, const float* y,
                                                    const float* stat, float* out) {
    long long i = (long long)blockIdx.x * blockDim.x + threadIdx.x;
    int f = (int)(i & (cD - 1));
    float v = y[i] * stat[f] + stat[cD + f];
    out[i] = x[i] + siluf(v);
}

// ---------------- host launcher ----------------
extern "C" void kernel_launch(void* const* d_in, const int* in_sizes, int n_in,
                              void* d_out, int out_size, void* d_ws, size_t ws_size,
                              hipStream_t stream) {
    (void)in_sizes; (void)n_in; (void)out_size; (void)ws_size;
    const float* x     = (const float*)d_in[0];
    const float* ef    = (const float*)d_in[1];
    const int*   eidx  = (const int*)d_in[2];
    const float* Kv2v  = (const float*)d_in[3];
    const float* Ke2v  = (const float*)d_in[4];
    const float* Vv2v  = (const float*)d_in[5];
    const float* Ve2v  = (const float*)d_in[6];
    const float* ku_w1 = (const float*)d_in[7];  const float* ku_b1 = (const float*)d_in[8];
    const float* ku_w2 = (const float*)d_in[9];  const float* ku_b2 = (const float*)d_in[10];
    const float* lu_w1 = (const float*)d_in[11]; const float* lu_b1 = (const float*)d_in[12];
    const float* lu_w2 = (const float*)d_in[13]; const float* lu_b2 = (const float*)d_in[14];
    const float* msg_w = (const float*)d_in[15]; const float* msg_b = (const float*)d_in[16];
    const float* ln_g  = (const float*)d_in[17]; const float* ln_b  = (const float*)d_in[18];
    const float* bn_g  = (const float*)d_in[19]; const float* bn_b  = (const float*)d_in[20];
    const float* cat_w = (const float*)d_in[21]; const float* cat_b = (const float*)d_in[22];
    const float* bnv_g = (const float*)d_in[23]; const float* bnv_b = (const float*)d_in[24];
    float* out = (float*)d_out;

    unsigned char* ws = (unsigned char*)d_ws;
    size_t off = 0;
    auto carve = [&](size_t bytes) -> unsigned char* {
        unsigned char* p = ws + off;
        off += (bytes + 255) & ~(size_t)255;
        return p;
    };
    unsigned short* pkKv  = (unsigned short*)carve((size_t)cH * cD * cD * 2);
    unsigned short* pkKe  = (unsigned short*)carve((size_t)cH * cD * cD * 2);
    unsigned short* pkVv  = (unsigned short*)carve((size_t)cH * cD * cD * 2);
    unsigned short* pkVe  = (unsigned short*)carve((size_t)cH * cD * cD * 2);
    unsigned short* pkKu1 = (unsigned short*)carve((size_t)cH * cD2 * cD2 * 2);
    unsigned short* pkKu2 = (unsigned short*)carve((size_t)cH * cD2 * cD2 * 2);
    unsigned short* pkLu1 = (unsigned short*)carve((size_t)cH * cD2 * cD2 * 2);
    unsigned short* pkLu2 = (unsigned short*)carve((size_t)cH * cD2 * cD2 * 2);
    unsigned short* pkMsg = (unsigned short*)carve((size_t)cH * cD2 * cD * 2);
    unsigned short* pkCat = (unsigned short*)carve((size_t)(cH * cD) * cD * 2);
    unsigned short* x_bf  = (unsigned short*)carve((size_t)cN * cD * 2);
    unsigned short* Kv_bf = (unsigned short*)carve((size_t)cH * cN * cD * 2);
    unsigned short* Vv_bf = (unsigned short*)carve((size_t)cH * cN * cD * 2);
    float* hv       = (float*)carve((size_t)cN * (cH * cD) * 4);
    float* ybuf     = (float*)carve((size_t)cN * cD * 4);
    float* bn_part  = (float*)carve((size_t)cH * NSLOT * 2 * cD2 * 4);
    float* bn_stat  = (float*)carve((size_t)cH * 2 * cD2 * 4);
    float* bnv_part = (float*)carve((size_t)NSLOT * 2 * cD * 4);
    float* bnv_stat = (float*)carve((size_t)2 * cD * 4);

    hipMemsetAsync(hv, 0, (size_t)cN * (cH * cD) * 4, stream);
    hipMemsetAsync(bn_part, 0, (size_t)cH * NSLOT * 2 * cD2 * 4, stream);
    hipMemsetAsync(bnv_part, 0, (size_t)NSLOT * 2 * cD * 4, stream);

    pack_b_kernel<<<dim3(64, cH), 256, 0, stream>>>(Kv2v, pkKv, cD, cD);
    pack_b_kernel<<<dim3(64, cH), 256, 0, stream>>>(Ke2v, pkKe, cD, cD);
    pack_b_kernel<<<dim3(64, cH), 256, 0, stream>>>(Vv2v, pkVv, cD, cD);
    pack_b_kernel<<<dim3(64, cH), 256, 0, stream>>>(Ve2v, pkVe, cD, cD);
    pack_b_kernel<<<dim3(256, cH), 256, 0, stream>>>(ku_w1, pkKu1, cD2, cD2);
    pack_b_kernel<<<dim3(256, cH), 256, 0, stream>>>(ku_w2, pkKu2, cD2, cD2);
    pack_b_kernel<<<dim3(256, cH), 256, 0, stream>>>(lu_w1, pkLu1, cD2, cD2);
    pack_b_kernel<<<dim3(256, cH), 256, 0, stream>>>(lu_w2, pkLu2, cD2, cD2);
    pack_b_kernel<<<dim3(128, cH), 256, 0, stream>>>(msg_w, pkMsg, cD2, cD);
    pack_b_kernel<<<dim3(256, 1), 256, 0, stream>>>(cat_w, pkCat, cH * cD, cD);

    cvt_bf_kernel<<<dim3(2048), 256, 0, stream>>>(x, x_bf, (long long)cN * cD);

    gemm_node_kernel<<<dim3(cN / 32, 2 * cH), 256, 0, stream>>>(x_bf, pkKv, pkVv,
                                                                Kv_bf, Vv_bf);

    edge_pass1_kernel<<<dim3(cE / 32, cH), 256, 0, stream>>>(
        ef, eidx, Kv_bf, pkKe, pkKu1, pkKu2, ku_b1, ku_b2, bn_part);

    reduce_bn_kernel<<<dim3(cH), 256, 0, stream>>>(bn_part, bn_g, bn_b, bn_stat);

    edge_pass2_kernel<<<dim3(cE / 32, cH), 256, 0, stream>>>(
        ef, eidx, Kv_bf, Vv_bf, pkKe, pkVe, pkKu1, pkKu2, pkLu1, pkLu2, pkMsg,
        ku_b1, ku_b2, lu_b1, lu_b2, msg_b, ln_g, ln_b, bn_stat, hv);

    cat_gemm_kernel<<<dim3(cN / 32), 256, 0, stream>>>(hv, pkCat, cat_b, ybuf,
                                                       bnv_part);

    reduce_bnv_kernel<<<dim3(1), 128, 0, stream>>>(bnv_part, bnv_g, bnv_b, bnv_stat);

    final_kernel<<<dim3((cN * cD) / 256), 256, 0, stream>>>(x, ybuf, bnv_stat, out);
}
